// MultiHeadAttentionBlock_29420525978120
// MI455X (gfx1250) — matrix operand order repaired and checked by reading, not compiled, not run on hardware
//
#include <hip/hip_runtime.h>

#define SEQ 4096
#define DM  1024
#define NH  16
#define DKH 64
#define LOG2E 1.4426950408889634f

typedef __bf16 bf16;
typedef __attribute__((ext_vector_type(16))) __bf16 v16bf;
typedef __attribute__((ext_vector_type(8)))  __bf16 v8bf;
typedef __attribute__((ext_vector_type(4)))  __bf16 v4bf;
typedef __attribute__((ext_vector_type(2)))  __bf16 v2bf;
typedef __attribute__((ext_vector_type(2)))  float  v2f;
typedef __attribute__((ext_vector_type(4)))  float  v4f;
typedef __attribute__((ext_vector_type(8)))  float  v8f;

// ---------------- fragment loaders (wave32, 16x16x32 bf16 WMMA) -------------
// A 16x32 (MxK): lane m=lane&15 holds row M; g=lane>>4: elems 0..7 = K g*8..,
// elems 8..15 = K 16+g*8..
__device__ __forceinline__ v16bf load_a32(const bf16* __restrict__ base, int ld) {
  const int lane = threadIdx.x & 31;
  const int m = lane & 15, g = lane >> 4;
  const bf16* p = base + m * ld + g * 8;
  v8bf lo = *(const v8bf*)(p);
  v8bf hi = *(const v8bf*)(p + 16);
  v16bf a;
#pragma unroll
  for (int i = 0; i < 8; ++i) { a[i] = lo[i]; a[8 + i] = hi[i]; }
  return a;
}

// B 32x16 (KxN): lane n=lane&15 holds column N; elems 0..7 = K g*8.., elems 8..15 = K 16+g*8..
// Column n lives at base + n*ld, K contiguous per lane.
__device__ __forceinline__ v16bf load_b32(const bf16* __restrict__ base, int ld) {
  const int lane = threadIdx.x & 31;
  const int n = lane & 15, g = lane >> 4;
  const bf16* p = base + (size_t)n * ld + g * 8;
  v8bf lo = *(const v8bf*)(p);
  v8bf hi = *(const v8bf*)(p + 16);
  v16bf b;
#pragma unroll
  for (int i = 0; i < 8; ++i) { b[i] = lo[i]; b[8 + i] = hi[i]; }
  return b;
}

__device__ __forceinline__ v4bf cvt4(v4f v) {
  v4bf r; r[0] = (bf16)v[0]; r[1] = (bf16)v[1]; r[2] = (bf16)v[2]; r[3] = (bf16)v[3];
  return r;
}

// ---------------- kernel 1: fused QKV projection (x @ W^T + b) --------------
// block = 256 thr = 8 waves; tile 128(s) x 128(d); K staged 64-wide thru LDS.
#define LDT 72  // padded LDS row stride (bf16) for 64-wide tiles

__global__ __launch_bounds__(256) void qkv_gemm_kernel(
    const float* __restrict__ x,
    const float* __restrict__ wq, const float* __restrict__ bq,
    const float* __restrict__ wk, const float* __restrict__ bk,
    const float* __restrict__ wv, const float* __restrict__ bv,
    bf16* __restrict__ Q, bf16* __restrict__ K, bf16* __restrict__ Vt) {
  const int mode = blockIdx.z;
  const float* __restrict__ w    = (mode == 0) ? wq : (mode == 1) ? wk : wv;
  const float* __restrict__ bias = (mode == 0) ? bq : (mode == 1) ? bk : bv;
  bf16* __restrict__ QK = (mode == 0) ? Q : K;

  const int sBase = blockIdx.x * 128;
  const int dBase = blockIdx.y * 128;

  __shared__ bf16 xs[128 * LDT];
  __shared__ bf16 wsT[128 * LDT];

  const int tid = threadIdx.x;
  const int wave = tid >> 5, lane = tid & 31;
  const int wm = wave >> 1, wn = wave & 1;   // 4 x 2 waves, 32(s) x 64(d) each

  v8f c[2][4] = {};

  for (int kc = 0; kc < DM; kc += 64) {
    // stage x and w tiles: float4 loads -> packed v4bf (8B) DS stores
#pragma unroll
    for (int i = 0; i < 8; ++i) {
      int p = tid + i * 256;            // 2048 quads = 128 x 16
      int r = p >> 4, qc = (p & 15) * 4;
      v4f xv = *(const v4f*)&x[(size_t)(sBase + r) * DM + kc + qc];
      *(v4bf*)&xs[r * LDT + qc] = cvt4(xv);
    }
#pragma unroll
    for (int i = 0; i < 8; ++i) {
      int p = tid + i * 256;
      int r = p >> 4, qc = (p & 15) * 4;
      v4f wv4 = *(const v4f*)&w[(size_t)(dBase + r) * DM + kc + qc];
      *(v4bf*)&wsT[r * LDT + qc] = cvt4(wv4);
    }
    __syncthreads();

#pragma unroll
    for (int cc = 0; cc < 2; ++cc) {
      v16bf a[2], b[4];
#pragma unroll
      for (int mi = 0; mi < 2; ++mi)
        a[mi] = load_a32(&xs[(wm * 32 + mi * 16) * LDT + cc * 32], LDT);
#pragma unroll
      for (int ni = 0; ni < 4; ++ni)
        b[ni] = load_b32(&wsT[(wn * 64 + ni * 16) * LDT + cc * 32], LDT);
#pragma unroll
      for (int mi = 0; mi < 2; ++mi)
#pragma unroll
        for (int ni = 0; ni < 4; ++ni)
          c[mi][ni] = __builtin_amdgcn_wmma_f32_16x16x32_bf16(
              false, a[mi], false, b[ni], (short)0, c[mi][ni], false, false);
    }
    __syncthreads();
  }

  const int g = lane >> 4, n = lane & 15;
  if (mode == 2) {
    // V transposed: for fixed lane, the 8 frag rows are contiguous in s ->
    // one packed 16B store per C-frag column.
#pragma unroll
    for (int mi = 0; mi < 2; ++mi)
#pragma unroll
      for (int ni = 0; ni < 4; ++ni) {
        const int d = dBase + wn * 64 + ni * 16 + n;
        const float bv = bias[d];
        v8bf pk;
#pragma unroll
        for (int r = 0; r < 8; ++r) pk[r] = (bf16)(c[mi][ni][r] + bv);
        const int s0 = sBase + wm * 32 + mi * 16 + 8 * g;
        *(v8bf*)&Vt[(size_t)d * SEQ + s0] = pk;
      }
  } else {
#pragma unroll
    for (int mi = 0; mi < 2; ++mi)
#pragma unroll
      for (int ni = 0; ni < 4; ++ni) {
        const int d = dBase + wn * 64 + ni * 16 + n;
        const float bv = bias[d];
#pragma unroll
        for (int r = 0; r < 8; ++r) {
          int s = sBase + wm * 32 + mi * 16 + r + 8 * g;
          QK[(size_t)s * DM + d] = (bf16)(c[mi][ni][r] + bv);
        }
      }
  }
}

// ---------------- kernel 2: flash attention with additive sim bias ----------
// block = 512 thr = 16 waves; wave w == head w; block handles 16 query rows.
// 64-key chunks; sim tile read ONCE per block, shared by all heads via LDS,
// double-buffered (one barrier per chunk). Softmax in base-2 domain.
#define SIMP 68   // padded f32 stride
#define LDPP 72   // padded P scratch stride (bf16)

__global__ __launch_bounds__(512) void attn_kernel(
    const bf16* __restrict__ Q, const bf16* __restrict__ K,
    const bf16* __restrict__ Vt, const float* __restrict__ sim,
    bf16* __restrict__ O) {
  const int rowBase = blockIdx.x * 16;
  const int tid = threadIdx.x;
  const int hd = tid >> 5;        // head == wave id
  const int lane = tid & 31;
  const int g = lane >> 4, n = lane & 15;

  __shared__ float simT[2][16 * SIMP];
  __shared__ bf16  pbuf[NH * 16 * LDPP];

  v16bf aq[2];
#pragma unroll
  for (int cc = 0; cc < 2; ++cc)
    aq[cc] = load_a32(Q + (size_t)rowBase * DM + hd * DKH + cc * 32, DM);

  v8f acc[4] = {};
  float mrow[8], lrow[8];
#pragma unroll
  for (int r = 0; r < 8; ++r) { mrow[r] = -3.0e38f; lrow[r] = 0.0f; }

  bf16* pmine = &pbuf[hd * 16 * LDPP];
  const float QK_SCALE = 0.125f * LOG2E;  // 1/sqrt(64) * log2(e)

  for (int kb = 0; kb < SEQ; kb += 64) {
    const int buf = (kb >> 6) & 1;
    // stage 16x64 sim tile as f32 pairs, pre-scaled by log2e (shared by heads)
    {
      int r = tid >> 5, pc = (tid & 31) * 2;   // 512 threads -> 512 pairs
      v2f sv = *(const v2f*)&sim[(size_t)(rowBase + r) * SEQ + kb + pc];
      sv[0] *= LOG2E; sv[1] *= LOG2E;
      *(v2f*)&simT[buf][r * SIMP + pc] = sv;
    }
    __syncthreads();
    if (kb + 64 < SEQ) {                // global_prefetch_b8 for next tile
      int r = tid >> 5, col = tid & 31;
      __builtin_prefetch(&sim[(size_t)(rowBase + r) * SEQ + kb + 64 + col * 2], 0, 0);
    }

    // scores: 16 rows x 64 keys; column mapping col = 2n + (nc&1) + 32*(nc>>1)
    // so each lane owns adjacent key pairs (packed P stores); key == kb + col.
    v8f sc[4];
#pragma unroll
    for (int nc = 0; nc < 4; ++nc) {
      const int colOff = (nc & 1) + 32 * (nc >> 1);
      v8f s = {};
#pragma unroll
      for (int cc = 0; cc < 2; ++cc) {
        v16bf b = load_b32(K + (size_t)(kb + colOff) * DM + hd * DKH + cc * 32,
                           2 * DM);     // lane n -> key kb + 2n + colOff
        s = __builtin_amdgcn_wmma_f32_16x16x32_bf16(
            false, aq[cc], false, b, (short)0, s, false, false);
      }
#pragma unroll
      for (int r = 0; r < 8; ++r) {
        int M = r + 8 * g;
        s[r] = fmaf(s[r], QK_SCALE, simT[buf][M * SIMP + 2 * n + colOff]);
      }
      sc[nc] = s;
    }

    // online softmax (base-2); row stats in the 16-lane group owning the row
#pragma unroll
    for (int r = 0; r < 8; ++r) {
      float t = fmaxf(fmaxf(sc[0][r], sc[1][r]), fmaxf(sc[2][r], sc[3][r]));
#pragma unroll
      for (int off = 1; off < 16; off <<= 1) t = fmaxf(t, __shfl_xor(t, off, 32));
      float mnew  = fmaxf(mrow[r], t);
      float alpha = exp2f(mrow[r] - mnew);
      float p0 = exp2f(sc[0][r] - mnew);
      float p1 = exp2f(sc[1][r] - mnew);
      float p2 = exp2f(sc[2][r] - mnew);
      float p3 = exp2f(sc[3][r] - mnew);
      sc[0][r] = p0; sc[1][r] = p1; sc[2][r] = p2; sc[3][r] = p3;
      float rs = (p0 + p1) + (p2 + p3);
#pragma unroll
      for (int off = 1; off < 16; off <<= 1) rs += __shfl_xor(rs, off, 32);
      lrow[r] = lrow[r] * alpha + rs;
      mrow[r] = mnew;
#pragma unroll
      for (int q = 0; q < 4; ++q) acc[q][r] *= alpha;
    }

    // P: C-frag -> LDS (packed b32 stores) -> A-frag
#pragma unroll
    for (int r = 0; r < 8; ++r) {
      int M = r + 8 * g;
      v2bf lo; lo[0] = (bf16)sc[0][r]; lo[1] = (bf16)sc[1][r];   // cols 2n,2n+1
      v2bf hi; hi[0] = (bf16)sc[2][r]; hi[1] = (bf16)sc[3][r];   // cols 32+2n..
      *(v2bf*)&pmine[M * LDPP + 2 * n]      = lo;
      *(v2bf*)&pmine[M * LDPP + 32 + 2 * n] = hi;
    }
    asm volatile("s_wait_dscnt 0" ::: "memory");
    v16bf ap0 = load_a32(pmine, LDPP);        // keys kb+0..31
    v16bf ap1 = load_a32(pmine + 32, LDPP);   // keys kb+32..63

    // O += P @ V  (Vt: two 16B key-contiguous B-frag loads per lane)
#pragma unroll
    for (int q = 0; q < 4; ++q) {
      const bf16* vb = Vt + (size_t)(hd * DKH + q * 16) * SEQ + kb;
      v16bf b0 = load_b32(vb, SEQ);
      acc[q] = __builtin_amdgcn_wmma_f32_16x16x32_bf16(
          false, ap0, false, b0, (short)0, acc[q], false, false);
      v16bf b1 = load_b32(vb + 32, SEQ);
      acc[q] = __builtin_amdgcn_wmma_f32_16x16x32_bf16(
          false, ap1, false, b1, (short)0, acc[q], false, false);
    }
    // no trailing barrier: double-buffered simT; next chunk's barrier protects
  }

  // normalize and emit O (bf16, [S][D] row-major for the output GEMM)
#pragma unroll
  for (int q = 0; q < 4; ++q)
#pragma unroll
    for (int r = 0; r < 8; ++r) {
      int s = rowBase + r + 8 * g;
      int d = hd * DKH + q * 16 + n;
      O[(size_t)s * DM + d] = (bf16)(acc[q][r] / lrow[r]);
    }
}

// ---------------- kernel 3: output projection (O @ Wo^T + bo), fp32 out -----
__global__ __launch_bounds__(256) void out_gemm_kernel(
    const bf16* __restrict__ O, const float* __restrict__ wo,
    const float* __restrict__ bo, float* __restrict__ out) {
  const int sBase = blockIdx.x * 128;
  const int dBase = blockIdx.y * 128;

  __shared__ bf16 xs[128 * LDT];
  __shared__ bf16 wsT[128 * LDT];

  const int tid = threadIdx.x;
  const int wave = tid >> 5, lane = tid & 31;
  const int wm = wave >> 1, wn = wave & 1;

  v8f c[2][4] = {};

  for (int kc = 0; kc < DM; kc += 64) {
    // O tile: pure 16B bf16 copies
#pragma unroll
    for (int i = 0; i < 4; ++i) {
      int p = tid + i * 256;            // 1024 octets = 128 x 8
      int r = p >> 3, qc = (p & 7) * 8;
      *(v8bf*)&xs[r * LDT + qc] =
          *(const v8bf*)&O[(size_t)(sBase + r) * DM + kc + qc];
    }
#pragma unroll
    for (int i = 0; i < 8; ++i) {
      int p = tid + i * 256;
      int r = p >> 4, qc = (p & 15) * 4;
      v4f wv4 = *(const v4f*)&wo[(size_t)(dBase + r) * DM + kc + qc];
      *(v4bf*)&wsT[r * LDT + qc] = cvt4(wv4);
    }
    __syncthreads();

#pragma unroll
    for (int cc = 0; cc < 2; ++cc) {
      v16bf a[2], b[4];
#pragma unroll
      for (int mi = 0; mi < 2; ++mi)
        a[mi] = load_a32(&xs[(wm * 32 + mi * 16) * LDT + cc * 32], LDT);
#pragma unroll
      for (int ni = 0; ni < 4; ++ni)
        b[ni] = load_b32(&wsT[(wn * 64 + ni * 16) * LDT + cc * 32], LDT);
#pragma unroll
      for (int mi = 0; mi < 2; ++mi)
#pragma unroll
        for (int ni = 0; ni < 4; ++ni)
          c[mi][ni] = __builtin_amdgcn_wmma_f32_16x16x32_bf16(
              false, a[mi], false, b[ni], (short)0, c[mi][ni], false, false);
    }
    __syncthreads();
  }

  const int g = lane >> 4, n = lane & 15;
#pragma unroll
  for (int mi = 0; mi < 2; ++mi)
#pragma unroll
    for (int ni = 0; ni < 4; ++ni) {
      const int d = dBase + wn * 64 + ni * 16 + n;
      const float bv = bo[d];
#pragma unroll
      for (int r = 0; r < 8; ++r) {
        int s = sBase + wm * 32 + mi * 16 + r + 8 * g;
        out[(size_t)s * DM + d] = c[mi][ni][r] + bv;
      }
    }
}

// ---------------- host launcher --------------------------------------------
extern "C" void kernel_launch(void* const* d_in, const int* in_sizes, int n_in,
                              void* d_out, int out_size, void* d_ws, size_t ws_size,
                              hipStream_t stream) {
  const float* x   = (const float*)d_in[0];
  const float* sim = (const float*)d_in[1];
  const float* wq  = (const float*)d_in[2];
  const float* bq  = (const float*)d_in[3];
  const float* wk  = (const float*)d_in[4];
  const float* bk  = (const float*)d_in[5];
  const float* wv  = (const float*)d_in[6];
  const float* bv  = (const float*)d_in[7];
  const float* wo  = (const float*)d_in[8];
  const float* bo  = (const float*)d_in[9];
  float* out = (float*)d_out;

  // workspace: Q,K bf16 [S][D]; Vt bf16 [D][S]; O bf16 [S][D]  (32 MB total)
  bf16* Q  = (bf16*)d_ws;
  bf16* K  = Q  + (size_t)SEQ * DM;
  bf16* Vt = K  + (size_t)SEQ * DM;
  bf16* O  = Vt + (size_t)DM * SEQ;

  qkv_gemm_kernel<<<dim3(SEQ / 128, DM / 128, 3), 256, 0, stream>>>(
      x, wq, bq, wk, bk, wv, bv, Q, K, Vt);

  attn_kernel<<<dim3(SEQ / 16), 512, 0, stream>>>(Q, K, Vt, sim, O);

  out_gemm_kernel<<<dim3(SEQ / 128, DM / 128), 256, 0, stream>>>(O, wo, bo, out);
}
